// BertCRF_1468878815447
// MI455X (gfx1250) — matrix operand order repaired and checked by reading, not compiled, not run on hardware
//
#include <hip/hip_runtime.h>

#define NUM_TAGS 9
#define HIDDEN   768
#define BATCH    64
#define SEQ      512
#define LP       510                 // L' = SEQ - 2
#define MROWS    (BATCH * LP)        // 32640 rows, = 2040 tiles of 16
#define KCHUNKS  (HIDDEN / 32)       // 24 WMMA K-steps
#define EMN      (LP * NUM_TAGS)     // 4590 emissions per batch
#define EMV4     ((EMN + 3) / 4)     // 1148 float4 async chunks

typedef __attribute__((ext_vector_type(16))) __bf16 v16bf;
typedef __attribute__((ext_vector_type(8)))  float  v8f;

// ---------------------------------------------------------------------------
// Kernel 0: build pre-swizzled bf16 B-fragments once (zero-padded tags 9->16).
// Layout: bfrag[(kc*32 + lane)*16 + e], in the contiguous K order also used for A:
//   col = lane&15, half = lane>>4, K = kc*32 + half*16 + e.
// 24 KB total -> L2-resident broadcast operand for every GEMM wave.
// ---------------------------------------------------------------------------
__global__ __launch_bounds__(256)
void bertcrf_prep_bfrag(const float* __restrict__ weight,
                        __bf16* __restrict__ bfrag)
{
    const int t = blockIdx.x * 256 + threadIdx.x;      // 0 .. 24*32-1
    if (t >= KCHUNKS * 32) return;
    const int kc   = t >> 5;
    const int lane = t & 31;
    const int col  = lane & 15;
    const int hi   = lane >> 4;
    const int kb   = kc * 32 + hi * 16;

    v16bf v;
    if (col < NUM_TAGS) {
        const float* w = weight + (size_t)col * HIDDEN + kb;
#pragma unroll
        for (int e = 0; e < 16; ++e) v[e] = (__bf16)w[e];
    } else {
#pragma unroll
        for (int e = 0; e < 16; ++e) v[e] = (__bf16)0.0f;
    }
    *(v16bf*)(bfrag + (size_t)t * 16) = v;
}

// ---------------------------------------------------------------------------
// Kernel 1: emissions[g, t] = hidden[b, l+1, :] . weight[t, :] + bias[t]
// One wave per 16-row tile; bf16 WMMA 16x16x32, f32 accumulate.
// Inner loop is branch-free: 4x B128 A-loads + 32B B-fragment + cvt + wmma.
// ---------------------------------------------------------------------------
__global__ __launch_bounds__(128)
void bertcrf_emis_wmma(const float* __restrict__ hidden,
                       const __bf16* __restrict__ bfrag,
                       const float* __restrict__ bias,
                       float* __restrict__ emis)
{
    const int wave = threadIdx.x >> 5;
    const int lane = threadIdx.x & 31;
    const int tile = blockIdx.x * 4 + wave;      // 2040 tiles total
    const int m    = lane & 15;                  // A row / B,C column index
    const int hi   = lane >> 4;                  // half-wave selector

    // This lane's A row (flattened (b,l) with the [1:-1] slice applied)
    const int g = tile * 16 + m;
    const int b = g / LP;
    const int l = g - b * LP;
    const float* arow = hidden + ((size_t)b * SEQ + (size_t)(l + 1)) * HIDDEN;

    const int   col = m;                         // output tag column (pad 9->16)
    const float bv  = (col < NUM_TAGS) ? bias[col] : 0.0f;
    const __bf16* bp = bfrag + (size_t)lane * 16;

    v8f c = {};

#pragma unroll 2
    for (int kc = 0; kc < KCHUNKS; ++kc) {
        const int kb = kc * 32;

        // A fragment (16x32 bf16), contiguous K order matching B:
        //  lanes 0-15 : elems 0..15 -> K=0..15
        //  lanes 16-31: elems 0..15 -> K=16..31
        const float4 a0 = *(const float4*)(arow + kb + hi * 16);
        const float4 a1 = *(const float4*)(arow + kb + hi * 16 + 4);
        const float4 a2 = *(const float4*)(arow + kb + hi * 16 + 8);
        const float4 a3 = *(const float4*)(arow + kb + hi * 16 + 12);
        v16bf a;
        a[0]=(__bf16)a0.x;  a[1]=(__bf16)a0.y;  a[2] =(__bf16)a0.z;  a[3] =(__bf16)a0.w;
        a[4]=(__bf16)a1.x;  a[5]=(__bf16)a1.y;  a[6] =(__bf16)a1.z;  a[7] =(__bf16)a1.w;
        a[8]=(__bf16)a2.x;  a[9]=(__bf16)a2.y;  a[10]=(__bf16)a2.z;  a[11]=(__bf16)a2.w;
        a[12]=(__bf16)a3.x; a[13]=(__bf16)a3.y; a[14]=(__bf16)a3.z;  a[15]=(__bf16)a3.w;

        // B fragment: pre-swizzled, zero-padded, L2-hot
        const v16bf bf = *(const v16bf*)(bp + (size_t)kc * (32 * 16));

        // (neg_a, A, neg_b, B, c_mod, C, reuse_a, reuse_b)
        c = __builtin_amdgcn_wmma_f32_16x16x32_bf16(false, a, false, bf,
                                                    (short)0, c, false, false);
    }

    // C layout: lane half selects M block (0..7 / 8..15), lane&15 = N
    if (col < NUM_TAGS) {
#pragma unroll
        for (int r = 0; r < 8; ++r) {
            const int gg = tile * 16 + hi * 8 + r;
            emis[(size_t)gg * NUM_TAGS + col] = c[r] + bv;
        }
    }
}

// ---------------------------------------------------------------------------
// Kernel 2: CRF numerator + log-partition (forward logsumexp) + Viterbi
// One wave per batch element; lane j owns tag j. Emissions staged into LDS
// with gfx1250 async global->LDS B128 transfers (ASYNCcnt-fenced), back-
// pointers as bytes in LDS, cross-tag reductions via wave shuffles.
// ---------------------------------------------------------------------------
__global__ __launch_bounds__(32)
void bertcrf_crf(const float* __restrict__ emis,
                 const int*   __restrict__ labels,
                 const float* __restrict__ start_t,
                 const float* __restrict__ end_t,
                 const float* __restrict__ trans,
                 float* __restrict__ partial,     // [BATCH]: num - den
                 float* __restrict__ tagsout)     // [BATCH][LP] floats
{
    __shared__ float em_s[EMN + 8];              // +pad for B128 tail
    __shared__ float trans_s[NUM_TAGS * NUM_TAGS];
    __shared__ unsigned char hist[(LP - 1) * 16];

    const int b = blockIdx.x;
    const int j = threadIdx.x;

    // Async-stage this batch's emissions into LDS (no VGPR round-trip).
    const float* gsrc = emis + (size_t)b * EMN;
    for (int i = j; i < EMV4; i += 32) {
        const unsigned ldsa = (unsigned)(uintptr_t)(&em_s[i * 4]);
        const unsigned long long ga =
            (unsigned long long)(uintptr_t)(gsrc + i * 4);
        asm volatile("global_load_async_to_lds_b128 %0, %1, off"
                     :: "v"(ldsa), "v"(ga) : "memory");
    }
    for (int i = j; i < NUM_TAGS * NUM_TAGS; i += 32)
        trans_s[i] = trans[i];
    asm volatile("s_wait_asynccnt 0x0" ::: "memory");
    __syncthreads();

    const int   jj   = (j < NUM_TAGS) ? j : (NUM_TAGS - 1);
    const float endj = end_t[jj];
    float tc[NUM_TAGS];                          // transition column into tag j
#pragma unroll
    for (int i = 0; i < NUM_TAGS; ++i) tc[i] = trans_s[i * NUM_TAGS + jj];

    const float NEGINF = -3.0e38f;
    float alpha = (j < NUM_TAGS) ? (start_t[jj] + em_s[j]) : NEGINF;
    float vsc   = alpha;

    const int lab0 = labels[b * SEQ + 1];
    const int tag0 = (lab0 >= 0) ? lab0 : 0;
    float score = start_t[tag0] + em_s[tag0];
    int   prev  = tag0;

    for (int l = 1; l < LP; ++l) {
        const float em_j = (j < NUM_TAGS) ? em_s[l * NUM_TAGS + j] : 0.0f;
        const int   lab  = labels[b * SEQ + l + 1];
        const bool  msk  = (lab >= 0);
        const int   tg   = msk ? lab : 0;

        // gather alpha_i and vscore_i from all tag lanes
        float va[NUM_TAGS], wa[NUM_TAGS];
#pragma unroll
        for (int i = 0; i < NUM_TAGS; ++i) {
            va[i] = __shfl(alpha, i, 32) + tc[i];
            wa[i] = __shfl(vsc,   i, 32) + tc[i];
        }
        // logsumexp over predecessors
        float mx = va[0];
#pragma unroll
        for (int i = 1; i < NUM_TAGS; ++i) mx = fmaxf(mx, va[i]);
        float s = 0.0f;
#pragma unroll
        for (int i = 0; i < NUM_TAGS; ++i) s += __expf(va[i] - mx);
        const float nxt = mx + __logf(s) + em_j;

        // max-plus + argmax (first-max tie break, matches jnp.argmax)
        float best = wa[0]; int bidx = 0;
#pragma unroll
        for (int i = 1; i < NUM_TAGS; ++i)
            if (wa[i] > best) { best = wa[i]; bidx = i; }
        const float nv = best + em_j;

        if (msk && j < NUM_TAGS) { alpha = nxt; vsc = nv; }
        if (j < 16) hist[(l - 1) * 16 + j] = (unsigned char)(msk ? bidx : j);

        // numerator path score (uniform across lanes)
        const float stp = trans_s[prev * NUM_TAGS + tg] + em_s[l * NUM_TAGS + tg];
        if (msk) { score += stp; prev = tg; }
    }

    // denominator = logsumexp_j(alpha_j + end_j)
    const float fj = (j < NUM_TAGS) ? (alpha + endj) : NEGINF;
    float fa[NUM_TAGS];
#pragma unroll
    for (int i = 0; i < NUM_TAGS; ++i) fa[i] = __shfl(fj, i, 32);
    float dmx = fa[0];
#pragma unroll
    for (int i = 1; i < NUM_TAGS; ++i) dmx = fmaxf(dmx, fa[i]);
    float ds = 0.0f;
#pragma unroll
    for (int i = 0; i < NUM_TAGS; ++i) ds += __expf(fa[i] - dmx);
    const float den = dmx + __logf(ds);

    // Viterbi terminal argmax
    const float gj = (j < NUM_TAGS) ? (vsc + endj) : NEGINF;
    float ga[NUM_TAGS];
#pragma unroll
    for (int i = 0; i < NUM_TAGS; ++i) ga[i] = __shfl(gj, i, 32);
    float gbest = ga[0]; int last = 0;
#pragma unroll
    for (int i = 1; i < NUM_TAGS; ++i)
        if (ga[i] > gbest) { gbest = ga[i]; last = i; }

    if (j == 0) {
        partial[b] = (score + end_t[prev]) - den;

        // backtrack: tags[l=s+1] = carry before consuming hist[s]
        int t = last;
        for (int s2 = LP - 2; s2 >= 0; --s2) {
            const int p = s2 + 1;
            const bool mp = labels[b * SEQ + p + 1] >= 0;
            tagsout[(size_t)b * LP + p] = mp ? (float)t : 0.0f;
            t = hist[s2 * 16 + t];
        }
        const bool m0 = labels[b * SEQ + 1] >= 0;
        tagsout[(size_t)b * LP + 0] = m0 ? (float)t : 0.0f;
    }
}

// ---------------------------------------------------------------------------
// Kernel 3: deterministic fixed-order reduction -> d_out[0] = -sum(llh_b)
// ---------------------------------------------------------------------------
__global__ void bertcrf_finalize(const float* __restrict__ partial,
                                 float* __restrict__ out0)
{
    if (threadIdx.x == 0 && blockIdx.x == 0) {
        float s = 0.0f;
        for (int i = 0; i < BATCH; ++i) s += partial[i];
        out0[0] = -s;
    }
}

extern "C" void kernel_launch(void* const* d_in, const int* in_sizes, int n_in,
                              void* d_out, int out_size, void* d_ws, size_t ws_size,
                              hipStream_t stream) {
    (void)in_sizes; (void)n_in; (void)out_size; (void)ws_size;
    const float* hidden  = (const float*)d_in[0];
    const int*   labels  = (const int*)  d_in[1];
    const float* weight  = (const float*)d_in[2];
    const float* bias    = (const float*)d_in[3];
    const float* start_t = (const float*)d_in[4];
    const float* end_t   = (const float*)d_in[5];
    const float* trans   = (const float*)d_in[6];

    float* out = (float*)d_out;                  // [0] = -llh, [1..] = tags

    // Workspace layout: [bfrag 24KB][emissions MROWS*9 f32][partial 64 f32]
    __bf16* ws_bf   = (__bf16*)d_ws;
    float*  ws_em   = (float*)((char*)d_ws + (size_t)KCHUNKS * 32 * 16 * sizeof(__bf16));
    float*  ws_part = ws_em + (size_t)MROWS * NUM_TAGS;

    bertcrf_prep_bfrag<<<3, 256, 0, stream>>>(weight, ws_bf);
    bertcrf_emis_wmma<<<MROWS / 16 / 4, 128, 0, stream>>>(hidden, ws_bf, bias, ws_em);
    bertcrf_crf<<<BATCH, 32, 0, stream>>>(ws_em, labels, start_t, end_t, trans,
                                          ws_part, out + 1);
    bertcrf_finalize<<<1, 32, 0, stream>>>(ws_part, out);
}
